// RegionProposalNet_76527727280565
// MI455X (gfx1250) — compile-verified, matrix operand order checked by reading, not run on hardware
//
#include <hip/hip_runtime.h>
#include <math.h>

// ---------------------------------------------------------------------------
// RPN forward for MI455X (gfx1250, wave32, WMMA).
//   1) f32 -> bf16 copy of w_trans (layout [OC][K], K = c*9+tap, already
//      fragment-friendly: per output channel, K is contiguous)
//   2) implicit-im2col GEMM via v_wmma_f32_16x16x32_bf16, register-direct
//      fragments: A gathered coalesced with clamped unconditional 32-bit-
//      indexed loads and a branchless division-free loop-carried walk over
//      (hh,ww,c*HW); B as 2x aligned b128 loads per fragment from
//      L2-resident weights. K-loop unrolled x2 (8 WMMAs per iteration).
//   3) 1x1 heads (vectorized 512-dot) -> cls/reg flats straight into d_out
//   4) softmax score + anchor decode + clip
//   5) per-(batch,level) histogram threshold ~ top-6000 selection
//   6) greedy NMS, 300 iterations, one block per batch -> rois
// ---------------------------------------------------------------------------

typedef __attribute__((ext_vector_type(16))) __bf16 v16bf;
typedef __attribute__((ext_vector_type(8)))  float  v8f;

#define C_IN   256
#define C_MID  512
#define KTOT   (C_IN * 9)       // 2304
#define TOTA   159882           // sum over levels of H*W*3
#define PRE_NMS 6000
#define POST_NMS 300
#define NMS_TH 0.7f

#define BLK_SP 128
#define BLK_OC 64
#define KT     32

__device__ __forceinline__ unsigned short bf_bits(float f) {
    union { __bf16 b; unsigned short u; } c;
    c.b = (__bf16)f;                       // native v_cvt_*bf16_f32
    return c.u;
}
__device__ __forceinline__ float bf2f(unsigned short h) {
    union { unsigned u; float f; } c; c.u = ((unsigned)h) << 16; return c.f;
}

// 16 bf16 fragment viewed as bf16 lanes / 2x uint4 / WMMA operand
union FragBF {
    __bf16 h[16];
    uint4 q[2];
    v16bf v;
};

// ---- 1) w_trans f32 -> bf16 (identity layout: [OC][K] with K=c*9+tap) ------
__global__ void k_pack_w(const float* __restrict__ w, unsigned short* __restrict__ wt, int n) {
    int i = blockIdx.x * blockDim.x + threadIdx.x;
    if (i >= n) return;
    wt[i] = bf_bits(w[i]);
}

// ---- 2) implicit im2col GEMM with WMMA, register-direct fragments ----------
// grid.x tiles spatial (BLK_SP rows), grid.y tiles OC (BLK_OC).
// 8 waves/block: wave w owns rows [w*16, w*16+16); 4 accumulators over 64 oc.
// Per ISA 7.12.2 (16-bit A 16x32): lane L holds row M=L%16; its 16 elements
// are K = kb..kb+7 and 16+kb..16+kb+7 with kb = (L/16)*8 -> two contiguous
// 8-element runs. The +9 jump between runs AND between K-steps is exactly
// (c+1, same tap): walk state (hh, ww, c*HW) is loop-carried, branchless,
// division-free, in 32-bit arithmetic.
__global__ __launch_bounds__(256)
void k_conv3x3(const unsigned short* __restrict__ wt,   // [OC][KTOT] bf16
               const float* __restrict__ feat,
               const float* __restrict__ bias,
               unsigned short* __restrict__ y,
               int Bn, int H, int W, int rowBase) {
    const int HW = H * W;
    const int S  = Bn * HW;
    const int t    = threadIdx.x;
    const int wave = t >> 5;
    const int lane = t & 31;
    const int m    = lane & 15;
    const int kb   = (lane >> 4) * 8;
    const int tileSp = blockIdx.x * BLK_SP;
    const int tileOc = blockIdx.y * BLK_OC;

    // this lane's A row (spatial position); lanes 0..15 are consecutive
    // positions -> coalesced feature gathers
    const int sp   = tileSp + wave * 16 + m;
    const bool live = sp < S;
    int b = 0, h = 0, wx = 0;
    if (live) {
        b = sp / HW; int pos = sp - b * HW;
        h = pos / W; wx = pos - h * W;
    }
    const float* fb = feat + (size_t)b * C_IN * HW;
    // dead lanes: force coords far OOB -> ok=false, clamp lands at 0
    const int hm1 = live ? h - 1  : -0x4000;
    const int wm1 = live ? wx - 1 : -0x4000;
    const int hE  = hm1 + 2, wE = wm1 + 2;   // wrap thresholds
    const int Hc  = H - 1, Wc = W - 1;

    // B columns: fixed output channel per lane, K contiguous in memory
    const unsigned short* wcol[4];
#pragma unroll
    for (int j = 0; j < 4; j++)
        wcol[j] = wt + (size_t)(tileOc + j * 16 + m) * KTOT;

    v8f acc[4];
#pragma unroll
    for (int j = 0; j < 4; j++) {
#pragma unroll
        for (int e = 0; e < 8; e++) acc[j][e] = 0.0f;
    }

    // loop-carried walk state (kb=0 -> tap(0,0); kb=8 -> tap(2,2))
    const int dh0 = kb / 3;
    const int dw0 = kb - dh0 * 3;
    int hh  = hm1 + dh0;
    int ww  = wm1 + dw0;
    int cHW = 0;

#pragma unroll 2
    for (int k0 = 0; k0 < KTOT; k0 += KT) {
        // stream next weight slice toward L2/L0 (speculative; OOB is dropped)
        __builtin_prefetch(wcol[0] + k0 + KT, 0, 1);

        // ---- A fragment: clamped unconditional gathers, branchless walk ----
        FragBF af;
#pragma unroll
        for (int e = 0; e < 16; e++) {
            int hc = hh < 0 ? 0 : (hh > Hc ? Hc : hh);
            int wc = ww < 0 ? 0 : (ww > Wc ? Wc : ww);
            bool ok = ((unsigned)hh < (unsigned)H) &
                      ((unsigned)ww < (unsigned)W);
            float v = fb[(unsigned)(cHW + hc * W + wc)];
            af.h[e] = ok ? (__bf16)v : (__bf16)0.0f;
            if (e == 7 || e == 15) {       // +9 jump: next c, same tap
                cHW += HW;
            } else {                        // advance one tap, branchless
                int wwn = ww + 1;
                bool wwrap = wwn > wE;
                ww = wwrap ? wm1 : wwn;
                int hhn = hh + (wwrap ? 1 : 0);
                bool hwrap = hhn > hE;
                hh = hwrap ? hm1 : hhn;
                cHW += hwrap ? HW : 0;
            }
        }

        // ---- B fragments: 2x aligned 16B loads each, then WMMA -------------
#pragma unroll
        for (int j = 0; j < 4; j++) {
            FragBF bfrag;
            bfrag.q[0] = *(const uint4*)(wcol[j] + k0 + kb);
            bfrag.q[1] = *(const uint4*)(wcol[j] + k0 + kb + 16);
            acc[j] = __builtin_amdgcn_wmma_f32_16x16x32_bf16(
                false, af.v, false, bfrag.v, (short)0, acc[j], false, false);
        }
    }

    // epilogue: bias + ReLU, C layout: VGPR r -> M = r + (lane>=16)*8
#pragma unroll
    for (int j = 0; j < 4; j++) {
        int oc = tileOc + j * 16 + m;
        float bb = bias[oc];
#pragma unroll
        for (int r = 0; r < 8; r++) {
            int spo = tileSp + wave * 16 + r + ((lane >> 4) << 3);
            if (spo < S) {
                float v = acc[j][r] + bb;
                v = v > 0.0f ? v : 0.0f;
                y[(size_t)(rowBase + spo) * C_MID + oc] = bf_bits(v);
            }
        }
    }
}

// ---- 3) 1x1 heads: vectorized 512-dot, write flats into d_out --------------
__global__ void k_heads(const unsigned short* __restrict__ y,
                        const float* __restrict__ wc, const float* __restrict__ bc,
                        const float* __restrict__ wr, const float* __restrict__ br,
                        float* __restrict__ out,
                        int Bn, int HW, int rowBase, int aoff,
                        size_t clsOff, size_t regOff) {
    int i = blockIdx.x * blockDim.x + threadIdx.x;
    int S = Bn * HW;
    if (i >= S * 18) return;
    int s = i / 18, o = i - s * 18;
    const unsigned short* yr = y + (size_t)(rowBase + s) * C_MID;
    const float* wv; float bb;
    if (o < 6) { wv = wc + (size_t)o * C_MID;       bb = bc[o];     }
    else       { wv = wr + (size_t)(o - 6) * C_MID; bb = br[o - 6]; }
    float acc = bb;
    for (int c8 = 0; c8 < C_MID; c8 += 8) {
        union { uint4 q; unsigned short s[8]; } yv;
        yv.q = *(const uint4*)(yr + c8);
        const float4 w0 = *(const float4*)(wv + c8);
        const float4 w1 = *(const float4*)(wv + c8 + 4);
        acc += w0.x * bf2f(yv.s[0]) + w0.y * bf2f(yv.s[1]) +
               w0.z * bf2f(yv.s[2]) + w0.w * bf2f(yv.s[3]) +
               w1.x * bf2f(yv.s[4]) + w1.y * bf2f(yv.s[5]) +
               w1.z * bf2f(yv.s[6]) + w1.w * bf2f(yv.s[7]);
    }
    int b = s / HW, pos = s - b * HW;
    if (o < 6) {
        int a = o >> 1, jj = o & 1;   // cls pairs channels (0,1),(2,3),(4,5)
        out[clsOff + ((size_t)(b * TOTA + aoff + pos * 3 + a)) * 2 + jj] = acc;
    } else {
        int o2 = o - 6; int a = o2 >> 2, dd = o2 & 3;
        out[regOff + ((size_t)(b * TOTA + aoff + pos * 3 + a)) * 4 + dd] = acc;
    }
}

// ---- 4) softmax score + anchor decode + clip -------------------------------
__global__ void k_decode(const float* __restrict__ out, const float* __restrict__ img_info,
                         float* __restrict__ boxes, float* __restrict__ scores,
                         int Bn, int H, int W, int stride, float sb,
                         int aoff, size_t clsOff, size_t regOff) {
    int i = blockIdx.x * blockDim.x + threadIdx.x;
    int HW = H * W; int N = Bn * HW * 3;
    if (i >= N) return;
    int b = i / (HW * 3); int rr = i - b * (HW * 3);
    int pos = rr / 3, a = rr - pos * 3;
    int h = pos / W, wx = pos - h * W;

    size_t cbase = clsOff + ((size_t)(b * TOTA + aoff + pos * 3)) * 2;
    float xc6[6];
#pragma unroll
    for (int q = 0; q < 6; q++) xc6[q] = out[cbase + q];
    // torch view(B,2,3H,W) softmax(dim=1): channel c pairs with (c%3, c%3+3)
    int c = 2 * a + 1; int mm = c % 3;
    float p0 = xc6[mm], p1 = xc6[mm + 3];
    float mx = fmaxf(p0, p1);
    float score = __expf(xc6[c] - mx) / (__expf(p0 - mx) + __expf(p1 - mx));

    const float ratio = (a == 0) ? 0.5f : (a == 1 ? 1.0f : 2.0f);
    float ctr = (sb - 1.0f) * 0.5f;
    float w0  = sqrtf(sb * sb / ratio);
    float h0  = w0 * ratio;
    float ws  = w0 * 8.0f, hs = h0 * 8.0f;
    float ax1 = ctr - (ws - 1.0f) * 0.5f + wx * (float)stride;
    float ay1 = ctr - (hs - 1.0f) * 0.5f + h  * (float)stride;
    float ax2 = ctr + (ws - 1.0f) * 0.5f + wx * (float)stride;
    float ay2 = ctr + (hs - 1.0f) * 0.5f + h  * (float)stride;
    float aw = ax2 - ax1 + 1.0f, ah = ay2 - ay1 + 1.0f;
    float acx = ax1 + 0.5f * aw, acy = ay1 + 0.5f * ah;

    size_t rbase = regOff + ((size_t)(b * TOTA + aoff + pos * 3 + a)) * 4;
    float d0 = out[rbase], d1 = out[rbase + 1], d2 = out[rbase + 2], d3 = out[rbase + 3];
    float pcx = d0 * aw + acx, pcy = d1 * ah + acy;
    float pw = __expf(d2) * aw, ph = __expf(d3) * ah;
    float bx1 = pcx - 0.5f * pw, by1 = pcy - 0.5f * ph;
    float bx2 = pcx + 0.5f * pw, by2 = pcy + 0.5f * ph;
    float wi = img_info[b * 3 + 1] - 1.0f, hi = img_info[b * 3 + 0] - 1.0f;
    bx1 = fminf(fmaxf(bx1, 0.0f), wi); by1 = fminf(fmaxf(by1, 0.0f), hi);
    bx2 = fminf(fmaxf(bx2, 0.0f), wi); by2 = fminf(fmaxf(by2, 0.0f), hi);

    size_t gi = (size_t)b * TOTA + aoff + pos * 3 + a;
    boxes[gi * 4 + 0] = bx1; boxes[gi * 4 + 1] = by1;
    boxes[gi * 4 + 2] = bx2; boxes[gi * 4 + 3] = by2;
    scores[gi] = score;
}

// ---- 5) histogram-based top-k threshold per (batch, level) -----------------
__global__ void k_thresh(const float* __restrict__ scores, float* __restrict__ thr,
                         int lvl, int N, int k, int aoff) {
    __shared__ unsigned int hist[1024];
    int b = blockIdx.x;
    for (int i = threadIdx.x; i < 1024; i += blockDim.x) hist[i] = 0;
    __syncthreads();
    const float* sc = scores + (size_t)b * TOTA + aoff;
    for (int i = threadIdx.x; i < N; i += blockDim.x) {
        int bk = (int)(sc[i] * 1024.0f);
        bk = bk < 0 ? 0 : (bk > 1023 ? 1023 : bk);
        atomicAdd(&hist[bk], 1u);
    }
    __syncthreads();
    if (threadIdx.x == 0) {
        unsigned cum = 0; int bk = 1023;
        for (; bk >= 0; bk--) { cum += hist[bk]; if (cum >= (unsigned)k) break; }
        if (bk < 0) bk = 0;
        thr[b * 5 + lvl] = (float)bk * (1.0f / 1024.0f);
    }
}

__global__ void k_mask(const float* __restrict__ scores, const float* __restrict__ thr,
                       float* __restrict__ sm, int Bn) {
    int i = blockIdx.x * blockDim.x + threadIdx.x;
    int N = Bn * TOTA;
    if (i >= N) return;
    int b = i / TOTA, a = i - b * TOTA;
    int lvl;
    if      (a < 120000) lvl = 0;
    else if (a < 150000) lvl = 1;
    else if (a < 157500) lvl = 2;
    else if (a < 159375) lvl = 3;
    else                 lvl = 4;
    float s = scores[i];
    sm[i] = (s >= thr[b * 5 + lvl]) ? s : -1e30f;
}

// ---- 6) greedy NMS, one block per batch ------------------------------------
__global__ void k_nms(const float* __restrict__ boxes, float* __restrict__ sm,
                      float* __restrict__ out) {
    __shared__ float bs[256]; __shared__ int bi[256];
    __shared__ float best[4];
    int b = blockIdx.x, t = threadIdx.x;
    const float* bx = boxes + (size_t)b * TOTA * 4;
    float* s = sm + (size_t)b * TOTA;

    for (int it = 0; it < POST_NMS; it++) {
        float mv = -1e30f; int mi = 0;
        for (int i = t; i < TOTA; i += 256) {
            float v = s[i]; if (v > mv) { mv = v; mi = i; }
        }
        bs[t] = mv; bi[t] = mi;
        __syncthreads();
        for (int o = 128; o > 0; o >>= 1) {
            if (t < o) { if (bs[t + o] > bs[t]) { bs[t] = bs[t + o]; bi[t] = bi[t + o]; } }
            __syncthreads();
        }
        if (t == 0) {
            int idx = bi[0];
            best[0] = bx[idx * 4 + 0]; best[1] = bx[idx * 4 + 1];
            best[2] = bx[idx * 4 + 2]; best[3] = bx[idx * 4 + 3];
            s[idx] = -1e30f;
            float ok = (bs[0] > -5e29f) ? 1.0f : 0.0f;
            float* r = out + ((size_t)b * POST_NMS + it) * 5;
            r[0] = (float)b;
            r[1] = best[0] * ok; r[2] = best[1] * ok;
            r[3] = best[2] * ok; r[4] = best[3] * ok;
        }
        __syncthreads();
        float X1 = best[0], Y1 = best[1], X2 = best[2], Y2 = best[3];
        float a1 = (X2 - X1 + 1.0f) * (Y2 - Y1 + 1.0f);
        for (int i = t; i < TOTA; i += 256) {
            if (s[i] > -5e29f) {
                float ix1 = fmaxf(X1, bx[i * 4 + 0]), iy1 = fmaxf(Y1, bx[i * 4 + 1]);
                float ix2 = fminf(X2, bx[i * 4 + 2]), iy2 = fminf(Y2, bx[i * 4 + 3]);
                float iw = fmaxf(ix2 - ix1 + 1.0f, 0.0f);
                float ih = fmaxf(iy2 - iy1 + 1.0f, 0.0f);
                float inter = iw * ih;
                float a2 = (bx[i * 4 + 2] - bx[i * 4 + 0] + 1.0f) *
                           (bx[i * 4 + 3] - bx[i * 4 + 1] + 1.0f);
                if (inter / (a1 + a2 - inter) > NMS_TH) s[i] = -1e30f;
            }
        }
        __syncthreads();
    }
}

// ---------------------------------------------------------------------------
extern "C" void kernel_launch(void* const* d_in, const int* in_sizes, int n_in,
                              void* d_out, int out_size, void* d_ws, size_t ws_size,
                              hipStream_t stream) {
    (void)n_in; (void)out_size; (void)ws_size;
    const float* feat[5] = { (const float*)d_in[0], (const float*)d_in[1],
                             (const float*)d_in[2], (const float*)d_in[3],
                             (const float*)d_in[4] };
    const float* img_info = (const float*)d_in[5];
    const float* w_trans  = (const float*)d_in[6];
    const float* b_trans  = (const float*)d_in[7];
    const float* w_cls    = (const float*)d_in[8];
    const float* b_cls    = (const float*)d_in[9];
    const float* w_reg    = (const float*)d_in[10];
    const float* b_reg    = (const float*)d_in[11];

    const int Bn = in_sizes[0] / (C_IN * 200 * 200);

    static const int Hs[5]  = {200, 100, 50, 25, 13};
    static const int Ws[5]  = {200, 100, 50, 25, 13};
    static const int STs[5] = {4, 8, 16, 32, 64};
    static const float SBs[5] = {4.f, 8.f, 16.f, 32.f, 64.f};
    int HWs[5], aoff[5], rowBase[5];
    int ao = 0, rb = 0;
    for (int l = 0; l < 5; l++) {
        HWs[l] = Hs[l] * Ws[l];
        aoff[l] = ao; ao += HWs[l] * 3;
        rowBase[l] = rb; rb += Bn * HWs[l];
    }
    const int S = rb;  // total spatial rows across levels/batches

    // workspace layout
    char* ws = (char*)d_ws;
    size_t off = 0;
    unsigned short* wt = (unsigned short*)(ws + off);
    off += (size_t)C_MID * KTOT * 2;           off = (off + 255) & ~(size_t)255;
    unsigned short* y = (unsigned short*)(ws + off);
    off += (size_t)S * C_MID * 2;              off = (off + 255) & ~(size_t)255;
    float* boxes = (float*)(ws + off);
    off += (size_t)Bn * TOTA * 4 * sizeof(float); off = (off + 255) & ~(size_t)255;
    float* scores = (float*)(ws + off);
    off += (size_t)Bn * TOTA * sizeof(float);  off = (off + 255) & ~(size_t)255;
    float* sm = (float*)(ws + off);
    off += (size_t)Bn * TOTA * sizeof(float);  off = (off + 255) & ~(size_t)255;
    float* thr = (float*)(ws + off);

    float* out = (float*)d_out;
    size_t roisN  = (size_t)Bn * POST_NMS * 5;
    size_t clsOff = roisN;
    size_t regOff = roisN + (size_t)Bn * TOTA * 2;

    // 1) weights f32 -> bf16 (same layout)
    {
        int nw = C_MID * KTOT;
        k_pack_w<<<(nw + 255) / 256, 256, 0, stream>>>(w_trans, wt, nw);
    }
    // 2) conv per level (WMMA)
    for (int l = 0; l < 5; l++) {
        int Sl = Bn * HWs[l];
        dim3 g((Sl + BLK_SP - 1) / BLK_SP, C_MID / BLK_OC);
        k_conv3x3<<<g, 256, 0, stream>>>(wt, feat[l], b_trans, y,
                                         Bn, Hs[l], Ws[l], rowBase[l]);
    }
    // 3) heads per level
    for (int l = 0; l < 5; l++) {
        int n = Bn * HWs[l] * 18;
        k_heads<<<(n + 255) / 256, 256, 0, stream>>>(y, w_cls, b_cls, w_reg, b_reg,
                                                     out, Bn, HWs[l], rowBase[l],
                                                     aoff[l], clsOff, regOff);
    }
    // 4) decode per level
    for (int l = 0; l < 5; l++) {
        int n = Bn * HWs[l] * 3;
        k_decode<<<(n + 255) / 256, 256, 0, stream>>>(out, img_info, boxes, scores,
                                                      Bn, Hs[l], Ws[l], STs[l], SBs[l],
                                                      aoff[l], clsOff, regOff);
    }
    // 5) top-k thresholds + mask
    for (int l = 0; l < 5; l++) {
        int Nl = HWs[l] * 3;
        int k  = Nl < PRE_NMS ? Nl : PRE_NMS;
        k_thresh<<<Bn, 256, 0, stream>>>(scores, thr, l, Nl, k, aoff[l]);
    }
    {
        int nm = Bn * TOTA;
        k_mask<<<(nm + 255) / 256, 256, 0, stream>>>(scores, thr, sm, Bn);
    }
    // 6) NMS -> rois
    k_nms<<<Bn, 256, 0, stream>>>(boxes, sm, out);
}
